// MSGGeneratorLSTM_49941879718578
// MI455X (gfx1250) — compile-verified, matrix operand order checked by reading, not run on hardware
//
#include <hip/hip_runtime.h>
#include <hip/hip_bf16.h>
#include <cstdint>
#include <cstddef>

// Problem sizes (match reference)
#define B_  4096
#define H_  1024
#define V_  64
#define L_  16

typedef __attribute__((ext_vector_type(16))) __bf16 v16bf;
typedef __attribute__((ext_vector_type(8)))  float  v8f;

union Frag { v16bf v; uint4 q[2]; };

__device__ __forceinline__ unsigned short f2bf(float f) {
    unsigned int u = __float_as_uint(f);
    u += 0x7fffu + ((u >> 16) & 1u);           // round-to-nearest-even
    return (unsigned short)(u >> 16);
}

// CDNA5 async global->LDS copy (tracked by ASYNCcnt, no VGPR staging)
__device__ __forceinline__ void async_g2l_b128(unsigned int lds_off, unsigned long long gaddr) {
    asm volatile("global_load_async_to_lds_b128 %0, %1, off"
                 :: "v"(lds_off), "v"(gaddr) : "memory");
}
__device__ __forceinline__ void wait_async0() {
    asm volatile("s_wait_asynccnt 0x0" ::: "memory");
}

// ---------------------------------------------------------------------------
// Prep kernels
// ---------------------------------------------------------------------------
__global__ void prep_w_fused(const float* __restrict__ W_ih,
                             const float* __restrict__ W_hh,
                             unsigned short* __restrict__ Bwt) {
    size_t idx = (size_t)blockIdx.x * blockDim.x + threadIdx.x;
    if (idx >= (size_t)4 * H_ * 2 * H_) return;
    int n = (int)(idx >> 11);
    int k = (int)(idx & 2047);
    float v = (k < H_) ? W_ih[(size_t)n * H_ + k]
                       : W_hh[(size_t)n * H_ + (k - H_)];
    Bwt[idx] = f2bf(v);
}

__global__ void prep_small(const float* __restrict__ W_out,
                           const float* __restrict__ b_ih,
                           const float* __restrict__ b_hh,
                           unsigned short* __restrict__ Woutb,
                           float* __restrict__ bcomb) {
    int idx = blockIdx.x * blockDim.x + threadIdx.x;
    if (idx < V_ * H_) Woutb[idx] = f2bf(W_out[idx]);
    if (idx < 4 * H_)  bcomb[idx] = b_ih[idx] + b_hh[idx];
}

// xh layout: [B][2H] bf16 : cols 0..H-1 = relu(x), cols H..2H-1 = h
__global__ void init_state(const float* __restrict__ enc_h,
                           const float* __restrict__ enc_c,
                           const float* __restrict__ init_input,
                           unsigned short* __restrict__ xh,
                           float* __restrict__ c,
                           float* __restrict__ mask) {
    int idx = blockIdx.x * blockDim.x + threadIdx.x;
    if (idx >= B_ * H_) return;
    int b = idx >> 10, j = idx & (H_ - 1);
    float x0 = init_input[j];
    xh[(size_t)b * 2 * H_ + j]      = f2bf(x0 > 0.0f ? x0 : 0.0f);
    xh[(size_t)b * 2 * H_ + H_ + j] = f2bf(enc_h[idx]);
    c[idx] = enc_c[idx];
    if (idx < L_ * B_) mask[idx] = 1.0f;
}

// ---------------------------------------------------------------------------
// bf16 GEMM with WMMA:  C[M][N] = A[M][K] * Bt[N][K]^T (+bias[n])
// Compile-time strides; double-buffered LDS filled by async global->LDS b128.
// 256 threads = 8 wave32 waves, block tile BM x BN, K-step 32.
// ---------------------------------------------------------------------------
template<int BM, int BN, int WCOLS, int SA, int SB, int SC, int K>
__global__ __launch_bounds__(256) void gemm_bf16_wmma(
    const unsigned short* __restrict__ A,
    const unsigned short* __restrict__ Bm,
    const float* __restrict__ bias,
    float* __restrict__ C)
{
    constexpr int KB    = 32;
    constexpr int LK    = KB + 8;            // padded LDS row (shorts) -> 80B rows
    constexpr int WROWS = 8 / WCOLS;
    constexpr int WM    = BM / WROWS;
    constexpr int WN    = BN / WCOLS;
    constexpr int MT    = WM / 16;
    constexpr int NT    = WN / 16;
    constexpr int NK    = K / KB;

    __shared__ unsigned short As[2][BM][LK];
    __shared__ unsigned short Bs[2][BN][LK];

    const int tid  = threadIdx.x;
    const int wid  = tid >> 5;
    const int lane = tid & 31;
    const int wm   = wid / WCOLS;
    const int wn   = wid % WCOLS;
    const int m16  = lane & 15;
    const int hf   = lane >> 4;

    const int rowBase = blockIdx.y * BM;
    const int colBase = blockIdx.x * BN;

    const unsigned short* Ab = A  + (size_t)rowBase * SA;
    const unsigned short* Bb = Bm + (size_t)colBase * SB;

    // issue async b128 copies for one K-tile into buffer `buf`
    auto issue = [&](int k0, int buf) {
        #pragma unroll
        for (int p = 0; p < (BM * 4) / 256; ++p) {
            int e = p * 256 + tid;
            int r = e >> 2, kk = (e & 3) * 8;
            async_g2l_b128((unsigned int)(uintptr_t)&As[buf][r][kk],
                           (unsigned long long)(uintptr_t)(Ab + (size_t)r * SA + k0 + kk));
        }
        #pragma unroll
        for (int p = 0; p < (BN * 4) / 256; ++p) {
            int e = p * 256 + tid;
            int r = e >> 2, kk = (e & 3) * 8;
            async_g2l_b128((unsigned int)(uintptr_t)&Bs[buf][r][kk],
                           (unsigned long long)(uintptr_t)(Bb + (size_t)r * SB + k0 + kk));
        }
    };

    v8f acc[MT][NT] = {};

    issue(0, 0);                               // prologue: fill buffer 0

    for (int kt = 0; kt < NK; ++kt) {
        const int cur = kt & 1;
        wait_async0();                         // my async copies into `cur` done
        __syncthreads();                       // everyone's copies done; everyone done reading cur^1
        if (kt + 1 < NK) issue((kt + 1) * KB, cur ^ 1);   // prefetch next tile

        // ---- fragment loads per ISA 16-bit A/B VGPR layout ----
        // lanes 0-15: row m16, K 0..7 & 16..23 ; lanes 16-31: row m16, K 8..15 & 24..31
        Frag a[MT], b[NT];
        #pragma unroll
        for (int mt = 0; mt < MT; ++mt) {
            int r = wm * WM + mt * 16 + m16;
            a[mt].q[0] = *(const uint4*)&As[cur][r][hf * 8];
            a[mt].q[1] = *(const uint4*)&As[cur][r][16 + hf * 8];
        }
        #pragma unroll
        for (int nt = 0; nt < NT; ++nt) {
            int r = wn * WN + nt * 16 + m16;
            b[nt].q[0] = *(const uint4*)&Bs[cur][r][hf * 8];
            b[nt].q[1] = *(const uint4*)&Bs[cur][r][16 + hf * 8];
        }

        #pragma unroll
        for (int mt = 0; mt < MT; ++mt)
            #pragma unroll
            for (int nt = 0; nt < NT; ++nt)
                acc[mt][nt] = __builtin_amdgcn_wmma_f32_16x16x32_bf16(
                    false, a[mt].v, false, b[nt].v,
                    (short)0, acc[mt][nt], false, false);
    }

    // ---- epilogue: C/D layout lanes 0-15 -> M=v, lanes 16-31 -> M=v+8 ----
    #pragma unroll
    for (int mt = 0; mt < MT; ++mt) {
        #pragma unroll
        for (int nt = 0; nt < NT; ++nt) {
            int col = colBase + wn * WN + nt * 16 + m16;
            float bv = bias ? bias[col] : 0.0f;
            int rb = rowBase + wm * WM + mt * 16 + hf * 8;
            #pragma unroll
            for (int v = 0; v < 8; ++v)
                C[(size_t)(rb + v) * SC + col] = acc[mt][nt][v] + bv;
        }
    }
}

// ---------------------------------------------------------------------------
// LSTM cell elementwise: torch gate order i,f,g,o; writes bf16 h into xh
// ---------------------------------------------------------------------------
__global__ void lstm_cell(const float* __restrict__ gates,
                          float* __restrict__ c,
                          unsigned short* __restrict__ xh) {
    int idx = blockIdx.x * blockDim.x + threadIdx.x;
    if (idx >= B_ * H_) return;
    int b = idx >> 10, j = idx & (H_ - 1);
    const float* gr = gates + (size_t)b * 4 * H_;
    float ig = gr[j], fg = gr[H_ + j], gg = gr[2 * H_ + j], og = gr[3 * H_ + j];
    float is = 1.0f / (1.0f + __expf(-ig));
    float fs = 1.0f / (1.0f + __expf(-fg));
    float os = 1.0f / (1.0f + __expf(-og));
    float gt = tanhf(gg);
    float cn = fs * c[idx] + is * gt;
    c[idx] = cn;
    float h = os * tanhf(cn);
    xh[(size_t)b * 2 * H_ + H_ + j] = f2bf(h);
}

// ---------------------------------------------------------------------------
// Softmax / argmax / one-hot / logp / embedding-gather (x = relu(emb[amax]))
// ---------------------------------------------------------------------------
__global__ void softmax_step(const float* __restrict__ logits,
                             float* __restrict__ msg,
                             float* __restrict__ logp,
                             const float* __restrict__ emb,
                             unsigned short* __restrict__ xh,
                             int first) {
    __shared__ float sl[V_];
    __shared__ int   s_am;
    __shared__ float s_lp;
    int b = blockIdx.x, tid = threadIdx.x;
    if (tid < V_) sl[tid] = logits[(size_t)b * V_ + tid];
    __syncthreads();
    if (tid == 0) {
        float mx = sl[0]; int am = 0;
        for (int v = 1; v < V_; ++v) if (sl[v] > mx) { mx = sl[v]; am = v; }
        float s = 0.0f;
        for (int v = 0; v < V_; ++v) s += __expf(sl[v] - mx);
        s_am = am;
        s_lp = -__logf(s);    // log(softmax at argmax) = -(log sum exp(l - max))
    }
    __syncthreads();
    if (tid < V_) msg[(size_t)b * V_ + tid] = (tid == s_am) ? 1.0f : 0.0f;
    if (tid == 0) logp[b] = (first ? 0.0f : logp[b]) + s_lp;
    const float* er = emb + (size_t)s_am * H_;
    for (int j = tid; j < H_; j += 256) {
        float x = er[j];
        xh[(size_t)b * 2 * H_ + j] = f2bf(x > 0.0f ? x : 0.0f);
    }
}

// ---------------------------------------------------------------------------
extern "C" void kernel_launch(void* const* d_in, const int* in_sizes, int n_in,
                              void* d_out, int out_size, void* d_ws, size_t ws_size,
                              hipStream_t stream) {
    const float* enc_h  = (const float*)d_in[0];
    const float* enc_c  = (const float*)d_in[1];
    const float* W_ih   = (const float*)d_in[2];
    const float* W_hh   = (const float*)d_in[3];
    const float* b_ih   = (const float*)d_in[4];
    const float* b_hh   = (const float*)d_in[5];
    const float* W_out  = (const float*)d_in[6];
    const float* b_out  = (const float*)d_in[7];
    const float* init_i = (const float*)d_in[8];
    const float* emb    = (const float*)d_in[9];

    float* out      = (float*)d_out;
    float* out_msg  = out;                                   // L*B*V
    float* out_mask = out_msg + (size_t)L_ * B_ * V_;        // L*1*B
    float* out_dig  = out_mask + (size_t)L_ * B_;            // L*B*V
    float* out_logp = out_dig + (size_t)L_ * B_ * V_;        // B

    char* ws = (char*)d_ws;
    unsigned short* xh    = (unsigned short*)ws; ws += (size_t)B_ * 2 * H_ * 2;
    unsigned short* Bwt   = (unsigned short*)ws; ws += (size_t)4 * H_ * 2 * H_ * 2;
    unsigned short* Woutb = (unsigned short*)ws; ws += (size_t)V_ * H_ * 2;
    float* bcomb = (float*)ws;                   ws += (size_t)4 * H_ * 4;
    float* cbuf  = (float*)ws;                   ws += (size_t)B_ * H_ * 4;
    float* gates = (float*)ws;                   ws += (size_t)B_ * 4 * H_ * 4;

    // one-time prep (weights fit in L2: 16MB fused + 128KB out proj)
    {
        size_t n = (size_t)4 * H_ * 2 * H_;
        prep_w_fused<<<dim3((unsigned)((n + 255) / 256)), 256, 0, stream>>>(W_ih, W_hh, Bwt);
    }
    prep_small<<<(V_ * H_ + 255) / 256, 256, 0, stream>>>(W_out, b_ih, b_hh, Woutb, bcomb);
    init_state<<<(B_ * H_ + 255) / 256, 256, 0, stream>>>(enc_h, enc_c, init_i, xh, cbuf, out_mask);

    for (int t = 0; t < L_; ++t) {
        // gates = [relu(x) | h] @ [W_ih^T ; W_hh^T] + (b_ih + b_hh)
        // 128x256 block tile, 8 waves of 64x64 -> 16 WMMA per wave per K-step
        dim3 g1(4 * H_ / 256, B_ / 128);
        gemm_bf16_wmma<128, 256, 4, 2 * H_, 2 * H_, 4 * H_, 2 * H_>
            <<<g1, 256, 0, stream>>>(xh, Bwt, bcomb, gates);

        lstm_cell<<<(B_ * H_ + 255) / 256, 256, 0, stream>>>(gates, cbuf, xh);

        // logits = h @ W_out^T + b_out, written directly into digits[t]
        float* dig_t = out_dig + (size_t)t * B_ * V_;
        dim3 g2(V_ / 64, B_ / 128);
        gemm_bf16_wmma<128, 64, 2, 2 * H_, H_, V_, H_>
            <<<g2, 256, 0, stream>>>(xh + H_, Woutb, b_out, dig_t);

        float* msg_t = out_msg + (size_t)t * B_ * V_;
        softmax_step<<<B_, 256, 0, stream>>>(dig_t, msg_t, out_logp, emb, xh, t == 0);
    }
}
